// LSTM_43989055045931
// MI455X (gfx1250) — matrix operand order repaired and checked by reading, not compiled, not run on hardware
//
#include <hip/hip_runtime.h>
#include <hip/hip_bf16.h>

// ---------------------------------------------------------------------------
// Problem constants (from reference)
// ---------------------------------------------------------------------------
#define SEQ_LEN   256
#define BATCH     64
#define INPUT     128
#define HIDDEN    1024
#define G4H       (4 * HIDDEN)     // 4096
#define FC1_OUT   512
#define NCLS      3

typedef __attribute__((ext_vector_type(16))) __bf16 v16bf;
typedef __attribute__((ext_vector_type(8)))  __bf16 v8bf;
typedef __attribute__((ext_vector_type(8)))  float  v8f;

// ---------------------------------------------------------------------------
// WMMA fragment loaders, matching CDNA5 16-bit A/B VGPR striping
// (CDNA5 ISA §7.12.2).  Each fragment = two 16B global_load_b128.
// ---------------------------------------------------------------------------
// A: 16x32 (MxK) bf16.  lane L: M=L%16, g=L/16.
//   halves 0..7  -> K = g*8 + [0..8);  halves 8..15 -> K = 16 + g*8 + [0..8)
__device__ __forceinline__ v16bf load_a_frag(const __bf16* __restrict__ base,
                                             size_t ld, int lane, int kbase) {
    const int m = lane & 15, g = lane >> 4;
    const __bf16* p = base + (size_t)m * ld + (size_t)kbase + (size_t)g * 8;
    v8bf lo = *(const v8bf*)(p);
    v8bf hi = *(const v8bf*)(p + 16);
    return __builtin_shufflevector(lo, hi, 0, 1, 2, 3, 4, 5, 6, 7,
                                           8, 9, 10, 11, 12, 13, 14, 15);
}

// B: 32x16 (KxN) bf16, sourced from row-major W [N, ld] (i.e. W^T as KxN).
// lane L: N=L%16, g=L/16; halves 0..7 -> K = g*8 + [0..8); halves 8..15 -> K = 16 + g*8 + [0..8)
__device__ __forceinline__ v16bf load_b_frag(const __bf16* __restrict__ wrow0,
                                             size_t ld, int lane, int kbase) {
    const int n = lane & 15, g = lane >> 4;
    const __bf16* p = wrow0 + (size_t)n * ld + (size_t)kbase + (size_t)g * 8;
    v8bf lo = *(const v8bf*)(p);
    v8bf hi = *(const v8bf*)(p + 16);
    return __builtin_shufflevector(lo, hi, 0, 1, 2, 3, 4, 5, 6, 7,
                                           8, 9, 10, 11, 12, 13, 14, 15);
}

__device__ __forceinline__ v8f wmma_bf16(v16bf a, v16bf b, v8f c) {
    return __builtin_amdgcn_wmma_f32_16x16x32_bf16(
        /*neg_a=*/false, a, /*neg_b=*/false, b,
        /*c_mod=*/(short)0, c, /*reuse_a=*/false, /*reuse_b=*/false);
}

__device__ __forceinline__ float sigf(float x) {
    return 1.0f / (1.0f + __expf(-x));
}

// One GEMM phase: B fragment loaded once per K-chunk, reused across the four
// 16-row batch tiles (4 accumulators) -> 10 loads per 4 WMMAs.
__device__ __forceinline__ void gemm_phase(const __bf16* __restrict__ abase,
                                           size_t lda,
                                           const __bf16* __restrict__ brow,
                                           size_t ldb, int kcount,
                                           int lane, v8f acc[4]) {
    const __bf16* a0 = abase;
    const __bf16* a1 = abase + (size_t)16 * lda;
    const __bf16* a2 = abase + (size_t)32 * lda;
    const __bf16* a3 = abase + (size_t)48 * lda;
    for (int k = 0; k < kcount; k += 32) {
        v16bf b  = load_b_frag(brow, ldb, lane, k);
        v16bf x0 = load_a_frag(a0, lda, lane, k);
        acc[0] = wmma_bf16(x0, b, acc[0]);
        v16bf x1 = load_a_frag(a1, lda, lane, k);
        acc[1] = wmma_bf16(x1, b, acc[1]);
        v16bf x2 = load_a_frag(a2, lda, lane, k);
        acc[2] = wmma_bf16(x2, b, acc[2]);
        v16bf x3 = load_a_frag(a3, lda, lane, k);
        acc[3] = wmma_bf16(x3, b, acc[3]);
    }
}

// ---------------------------------------------------------------------------
// Per-(layer, t) step arguments
// ---------------------------------------------------------------------------
struct StepArgs {
    const __bf16* xin;     // A phase-1 base (row 0 of batch)
    long long     ldx;     // A phase-1 row stride (elements)
    int           kx;      // phase-1 K extent
    int           active;  // 0 -> this layer idle this launch
    const __bf16* hin;     // [B,H] bf16 h_{t-1}
    const __bf16* wih;     // [4H, kx] bf16
    const __bf16* whh;     // [4H, H]  bf16
    const float*  bias;    // [4H] = b_ih + b_hh
    float*        cst;     // [B,H] cell state (in/out)
    __bf16*       hout;    // [B,H] h_t
    __bf16*       hcopy;   // optional extra copy of h_t (final step)
};

// ---------------------------------------------------------------------------
// Fused, layer-pipelined LSTM step.  Launch tl runs L0(tl) and L1(tl-1)
// concurrently:  blocks 0..63 -> layer 0, 64..127 -> layer 1.
// Per block: j_tile = 16 hidden units; 4 waves = 4 gates; each wave computes
// its gate's 16x16 tiles for ALL four batch tiles via WMMA, then the block
// fuses the LSTM cell update through LDS.
// ---------------------------------------------------------------------------
__global__ __launch_bounds__(128)
void lstm_pair_kernel(StepArgs s0, StepArgs s1) {
    const StepArgs s = (blockIdx.x < 64) ? s0 : s1;
    if (!s.active) return;                    // block-uniform

    const int lane   = threadIdx.x & 31;
    const int gate   = threadIdx.x >> 5;      // wave id = gate 0..3 (i,f,g,o)
    const int j_tile = blockIdx.x & 63;       // hidden tile (1024/16)

    __shared__ float gtile[4][BATCH][17];     // pad 17 to dodge bank conflicts

    v8f acc[4];
#pragma unroll
    for (int m = 0; m < 4; ++m)
        acc[m] = (v8f){0.f, 0.f, 0.f, 0.f, 0.f, 0.f, 0.f, 0.f};

    const size_t jrow = (size_t)(gate * HIDDEN + j_tile * 16);
    // Phase 1: x_t @ W_ih^T    Phase 2: h_{t-1} @ W_hh^T
    gemm_phase(s.xin, (size_t)s.ldx, s.wih + jrow * (size_t)s.kx,
               (size_t)s.kx, s.kx, lane, acc);
    gemm_phase(s.hin, HIDDEN, s.whh + jrow * HIDDEN,
               HIDDEN, HIDDEN, lane, acc);

    // Spill accum tiles.  C/D layout: VGPR r -> M = r + 8*(lane/16); lane%16 -> N.
    {
        const int col = lane & 15, g = lane >> 4;
#pragma unroll
        for (int m = 0; m < 4; ++m)
#pragma unroll
            for (int r = 0; r < 8; ++r)
                gtile[gate][m * 16 + g * 8 + r][col] = acc[m][r];
    }
    __syncthreads();

    // Fused LSTM pointwise: 64 batch x 16 units = 1024 elems, 8 per thread.
    for (int e = threadIdx.x; e < BATCH * 16; e += 128) {
        const int row = e >> 4, cc = e & 15;       // batch row, unit-in-tile
        const int j   = j_tile * 16 + cc;          // hidden unit
        const float gi = gtile[0][row][cc] + s.bias[0 * HIDDEN + j];
        const float gf = gtile[1][row][cc] + s.bias[1 * HIDDEN + j];
        const float gg = gtile[2][row][cc] + s.bias[2 * HIDDEN + j];
        const float go = gtile[3][row][cc] + s.bias[3 * HIDDEN + j];
        const size_t idx = (size_t)row * HIDDEN + j;
        const float cn = sigf(gf) * s.cst[idx] + sigf(gi) * tanhf(gg);
        s.cst[idx] = cn;
        const float hn = sigf(go) * tanhf(cn);
        const __bf16 hb = (__bf16)hn;
        s.hout[idx] = hb;
        if (s.hcopy) s.hcopy[idx] = hb;
    }
}

// ---------------------------------------------------------------------------
// Generic bf16 WMMA GEMM with bias:  C[M,N] = A[M,K] @ Bw[N,K]^T + bias[N]
// One wave per 16x16 output tile (head is tiny; this is fine).
// ---------------------------------------------------------------------------
__global__ __launch_bounds__(128)
void gemm_bf16_kernel(const __bf16* __restrict__ A,
                      const __bf16* __restrict__ Bw,
                      const float*  __restrict__ bias,
                      float*        __restrict__ C,
                      int M, int N, int K)
{
    const int lane = threadIdx.x & 31;
    const int wv   = blockIdx.x * 4 + (threadIdx.x >> 5);
    const int n_tiles = N / 16;
    const int m_tile = wv / n_tiles;
    const int n_tile = wv % n_tiles;
    if (m_tile * 16 >= M) return;   // wave-uniform: EXEC stays all-ones

    v8f acc = {0.f, 0.f, 0.f, 0.f, 0.f, 0.f, 0.f, 0.f};
    const __bf16* arow = A  + (size_t)(m_tile * 16) * (size_t)K;
    const __bf16* brow = Bw + (size_t)(n_tile * 16) * (size_t)K;
    for (int k = 0; k < K; k += 32) {
        v16bf a = load_a_frag(arow, (size_t)K, lane, k);
        v16bf b = load_b_frag(brow, (size_t)K, lane, k);
        acc = wmma_bf16(a, b, acc);
    }
    const int col = lane & 15, g = lane >> 4;
    const float bv = bias[n_tile * 16 + col];
#pragma unroll
    for (int r = 0; r < 8; ++r) {
        const int m = m_tile * 16 + g * 8 + r;
        C[(size_t)m * N + n_tile * 16 + col] = acc[r] + bv;
    }
}

// ---------------------------------------------------------------------------
// Tiny kernels: f32 -> bf16 convert, bias sum, final 3-class projection.
// ---------------------------------------------------------------------------
__global__ void f2bf_kernel(const float* __restrict__ in,
                            __bf16* __restrict__ out, int n) {
    int i = blockIdx.x * blockDim.x + threadIdx.x;
    if (i < n) out[i] = (__bf16)in[i];
}

__global__ void bias_sum_kernel(const float* __restrict__ a,
                                const float* __restrict__ b,
                                float* __restrict__ out, int n) {
    int i = blockIdx.x * blockDim.x + threadIdx.x;
    if (i < n) out[i] = a[i] + b[i];
}

__global__ void head2_kernel(const float* __restrict__ z,     // [128, 512]
                             const float* __restrict__ fc2w,  // [3, 512]
                             const float* __restrict__ fc2b,  // [3]
                             float* __restrict__ out)         // [128, 3]
{
    const int idx = threadIdx.x;                 // 0..383
    if (idx >= 128 * NCLS) return;
    const int row = idx / NCLS, cls = idx % NCLS;
    const float* zr = z + (size_t)row * FC1_OUT;
    const float* wr = fc2w + (size_t)cls * FC1_OUT;
    float s = fc2b[cls];
    for (int k = 0; k < FC1_OUT; ++k) s = fmaf(zr[k], wr[k], s);
    out[row * NCLS + cls] = s;
}

// ---------------------------------------------------------------------------
// Host-side launch
// ---------------------------------------------------------------------------
extern "C" void kernel_launch(void* const* d_in, const int* in_sizes, int n_in,
                              void* d_out, int out_size, void* d_ws, size_t ws_size,
                              hipStream_t stream) {
    (void)in_sizes; (void)n_in; (void)out_size; (void)ws_size;
    const float* x     = (const float*)d_in[0];
    const float* w_ih0 = (const float*)d_in[1];
    const float* w_hh0 = (const float*)d_in[2];
    const float* b_ih0 = (const float*)d_in[3];
    const float* b_hh0 = (const float*)d_in[4];
    const float* w_ih1 = (const float*)d_in[5];
    const float* w_hh1 = (const float*)d_in[6];
    const float* b_ih1 = (const float*)d_in[7];
    const float* b_hh1 = (const float*)d_in[8];
    const float* fc1_w = (const float*)d_in[9];
    const float* fc1_b = (const float*)d_in[10];
    const float* fc2_w = (const float*)d_in[11];
    const float* fc2_b = (const float*)d_in[12];
    float* out = (float*)d_out;

    // ---- carve workspace (~67 MB) ----
    char* ws = (char*)d_ws;
    auto alloc = [&](size_t bytes) -> char* {
        char* p = ws; ws += (bytes + 255) & ~(size_t)255; return p;
    };
    __bf16* wih0b = (__bf16*)alloc((size_t)G4H * INPUT  * 2);
    __bf16* whh0b = (__bf16*)alloc((size_t)G4H * HIDDEN * 2);
    __bf16* wih1b = (__bf16*)alloc((size_t)G4H * HIDDEN * 2);
    __bf16* whh1b = (__bf16*)alloc((size_t)G4H * HIDDEN * 2);
    __bf16* xb    = (__bf16*)alloc((size_t)BATCH * SEQ_LEN * INPUT * 2);
    __bf16* ys0   = (__bf16*)alloc((size_t)SEQ_LEN * BATCH * HIDDEN * 2);
    __bf16* h1a   = (__bf16*)alloc((size_t)BATCH * HIDDEN * 2);
    __bf16* h1b   = (__bf16*)alloc((size_t)BATCH * HIDDEN * 2);
    __bf16* hz    = (__bf16*)alloc((size_t)BATCH * HIDDEN * 2);   // zero h
    __bf16* hfin  = (__bf16*)alloc((size_t)2 * BATCH * HIDDEN * 2);
    __bf16* fc1wb = (__bf16*)alloc((size_t)FC1_OUT * HIDDEN * 2);
    float*  bias0 = (float*)alloc((size_t)G4H * 4);
    float*  bias1 = (float*)alloc((size_t)G4H * 4);
    float*  c0    = (float*)alloc((size_t)BATCH * HIDDEN * 4);
    float*  c1    = (float*)alloc((size_t)BATCH * HIDDEN * 4);
    float*  zbuf  = (float*)alloc((size_t)2 * BATCH * FC1_OUT * 4);

    // ---- init state (every call: harness poisons ws once, never restores) --
    hipMemsetAsync(hz,  0, (size_t)BATCH * HIDDEN * 2, stream);
    hipMemsetAsync(h1a, 0, (size_t)BATCH * HIDDEN * 2, stream);
    hipMemsetAsync(c0,  0, (size_t)BATCH * HIDDEN * 4, stream);
    hipMemsetAsync(c1,  0, (size_t)BATCH * HIDDEN * 4, stream);

    // ---- convert weights / inputs to bf16; fold biases ----
    auto conv = [&](const float* src, __bf16* dst, int n) {
        f2bf_kernel<<<(n + 255) / 256, 256, 0, stream>>>(src, dst, n);
    };
    conv(w_ih0, wih0b, G4H * INPUT);
    conv(w_hh0, whh0b, G4H * HIDDEN);
    conv(w_ih1, wih1b, G4H * HIDDEN);
    conv(w_hh1, whh1b, G4H * HIDDEN);
    conv(x,     xb,    BATCH * SEQ_LEN * INPUT);
    conv(fc1_w, fc1wb, FC1_OUT * HIDDEN);
    bias_sum_kernel<<<(G4H + 255) / 256, 256, 0, stream>>>(b_ih0, b_hh0, bias0, G4H);
    bias_sum_kernel<<<(G4H + 255) / 256, 256, 0, stream>>>(b_ih1, b_hh1, bias1, G4H);

    // ---- recurrence: layer-pipelined, 257 launches ----
    const long long x_row_stride = (long long)SEQ_LEN * INPUT;  // x is [B,T,D]
    for (int tl = 0; tl <= SEQ_LEN; ++tl) {
        StepArgs s0{}, s1{};
        // Layer 0 runs step tl
        if (tl < SEQ_LEN) {
            const int t = tl;
            s0.xin  = xb + (size_t)t * INPUT;
            s0.ldx  = x_row_stride;
            s0.kx   = INPUT;
            s0.active = 1;
            s0.hin  = (t == 0) ? hz : ys0 + (size_t)(t - 1) * BATCH * HIDDEN;
            s0.wih  = wih0b;  s0.whh = whh0b;  s0.bias = bias0;  s0.cst = c0;
            s0.hout = ys0 + (size_t)t * BATCH * HIDDEN;
            s0.hcopy = (t == SEQ_LEN - 1) ? hfin : nullptr;
        }
        // Layer 1 runs step tl-1 (pipelined one launch behind)
        if (tl >= 1) {
            const int t = tl - 1;
            const __bf16* y0 = ys0 + (size_t)t * BATCH * HIDDEN;
            s1.xin  = y0;
            s1.ldx  = HIDDEN;
            s1.kx   = HIDDEN;
            s1.active = 1;
            s1.hin  = (t == 0) ? hz : ((t & 1) ? h1b : h1a);
            s1.wih  = wih1b;  s1.whh = whh1b;  s1.bias = bias1;  s1.cst = c1;
            s1.hout = (t & 1) ? h1a : h1b;
            s1.hcopy = (t == SEQ_LEN - 1) ? hfin + (size_t)BATCH * HIDDEN : nullptr;
        }
        lstm_pair_kernel<<<128, 128, 0, stream>>>(s0, s1);
    }

    // ---- MLP head ----
    // z = h_out @ fc1_w^T + fc1_b :  M=128, N=512, K=1024 -> 256 tile-waves
    gemm_bf16_kernel<<<64, 128, 0, stream>>>(hfin, fc1wb, fc1_b, zbuf,
                                             2 * BATCH, FC1_OUT, HIDDEN);
    // out = z @ fc2_w^T + fc2_b
    head2_kernel<<<1, 384, 0, stream>>>(zbuf, fc2_w, fc2_b, out);
}
